// CausalSelfAttention_32779190403632
// MI455X (gfx1250) — matrix operand order repaired and checked by reading, not compiled, not run on hardware
//
#include <hip/hip_runtime.h>

typedef __attribute__((ext_vector_type(4)))  __bf16 bf16x4;
typedef __attribute__((ext_vector_type(8)))  __bf16 bf16x8;
typedef __attribute__((ext_vector_type(16))) __bf16 bf16x16;
typedef __attribute__((ext_vector_type(8)))  float  floatx8;
typedef __attribute__((ext_vector_type(4)))  int    i32x4;

#define N_B   4
#define N_T   2048
#define N_D   768
#define N_H   12
#define N_HD  64
#define N_3D  2304
#define N_M   (N_B * N_T)   // 8192

// ---------------- async global->LDS (CDNA5), with safe fallback -------------
#if defined(__has_builtin)
#if __has_builtin(__builtin_amdgcn_global_load_async_to_lds_b128) && \
    __has_builtin(__builtin_amdgcn_s_wait_asynccnt)
#define HAS_ASYNC_LDS 1
#endif
#endif

static __device__ inline void copy16_g2s(__bf16* dst, const __bf16* src) {
#ifdef HAS_ASYNC_LDS
  __builtin_amdgcn_global_load_async_to_lds_b128(
      (__attribute__((address_space(1))) i32x4*)const_cast<__bf16*>(src),
      (__attribute__((address_space(3))) i32x4*)dst, 0, 0);
#else
  *(uint4*)dst = *(const uint4*)src;
#endif
}

#ifdef HAS_ASYNC_LDS
#define ASYNC_WAIT(n) __builtin_amdgcn_s_wait_asynccnt(n)
#else
#define ASYNC_WAIT(n) ((void)0)
#endif

// ---------------------------------------------------------------------------
static __device__ inline floatx8 wmma_bf16(bf16x16 a, bf16x16 b, floatx8 c) {
  return __builtin_amdgcn_wmma_f32_16x16x32_bf16(false, a, false, b, (short)0, c,
                                                 false, false);
}

// A fragment from row-major 16x32 tile (stride lda, multiple of 8 elements):
// lane L: row m=L&15, halves [kb..kb+7] and [16+kb..16+kb+7], kb = (L>=16)*8.
static __device__ inline bf16x16 frag_a(const __bf16* src, int lda, int lane) {
  const int m  = lane & 15;
  const int kb = (lane >> 4) << 3;
  const __bf16* p = src + m * lda + kb;
  bf16x8 lo = *(const bf16x8*)p;
  bf16x8 hi = *(const bf16x8*)(p + 16);
  return __builtin_shufflevector(lo, hi, 0, 1, 2, 3, 4, 5, 6, 7,
                                 8, 9, 10, 11, 12, 13, 14, 15);
}

// B fragment where B[k][n] is stored K-contiguous: srcT[n][k] (stride ldt,
// multiple of 8 elements). Lane L: n=L&15, halves [kb..kb+7] and [16+kb..16+kb+7], kb=(L>=16)*8.
static __device__ inline bf16x16 frag_b(const __bf16* srcT, int ldt, int lane) {
  const int n  = lane & 15;
  const int kb = (lane >> 4) << 3;
  const __bf16* p = srcT + n * ldt + kb;
  bf16x8 lo = *(const bf16x8*)p;
  bf16x8 hi = *(const bf16x8*)(p + 16);
  return __builtin_shufflevector(lo, hi, 0, 1, 2, 3, 4, 5, 6, 7,
                                 8, 9, 10, 11, 12, 13, 14, 15);
}

// ---------------------------------------------------------------------------
// Pre-pass 1: fp32 -> bf16 flat convert (x).
// ---------------------------------------------------------------------------
__global__ __launch_bounds__(256) void cvt_bf16_kernel(
    const float* __restrict__ src, __bf16* __restrict__ dst, int n4) {
  int i = blockIdx.x * 256 + threadIdx.x;
  if (i >= n4) return;
  float4 v = *(const float4*)(src + (size_t)i * 4);
  bf16x4 o;
  o[0] = (__bf16)v.x; o[1] = (__bf16)v.y; o[2] = (__bf16)v.z; o[3] = (__bf16)v.w;
  *(bf16x4*)(dst + (size_t)i * 4) = o;
}

// ---------------------------------------------------------------------------
// Pre-pass 2: transpose + convert weight W[K][N] fp32 -> Wt[N][K] bf16.
// ---------------------------------------------------------------------------
__global__ __launch_bounds__(256) void transpose_cvt_kernel(
    const float* __restrict__ W, __bf16* __restrict__ Wt, int K, int N) {
  __shared__ __align__(16) __bf16 tile[32][33];
  const int tid = threadIdx.x;
  const int k0 = blockIdx.y * 32, n0 = blockIdx.x * 32;
#pragma unroll
  for (int i = 0; i < 4; ++i) {
    int idx = tid + i * 256;
    int r = idx >> 5, c = idx & 31;
    tile[r][c] = (__bf16)W[(size_t)(k0 + r) * N + n0 + c];
  }
  __syncthreads();
#pragma unroll
  for (int i = 0; i < 4; ++i) {
    int idx = tid + i * 256;
    int r = idx >> 5, c = idx & 31;
    Wt[(size_t)(n0 + r) * K + k0 + c] = tile[c][r];
  }
}

// ---------------------------------------------------------------------------
// Kernel 1: qkv = xb @ Wqkvt^T + b_qkv, scattered to q/k [B,H,T,hd] and
// v [B,H,hd,T] (transposed) bf16. 128x128x32 tiles, 8 waves (2x4),
// double-buffered async global->LDS pipeline (in-order ASYNCcnt).
// ---------------------------------------------------------------------------
#define BT_M 128
#define BT_N 128
#define BT_K 32
#define NK_STEPS (N_D / BT_K)   // 24
#define LD_S 48                 // K-tile stride with pad, multiple of 8

__global__ __launch_bounds__(256) void qkv_gemm(
    const __bf16* __restrict__ xb, const __bf16* __restrict__ Wt,
    const float* __restrict__ bias, __bf16* __restrict__ wq,
    __bf16* __restrict__ wk, __bf16* __restrict__ wv) {
  __shared__ __align__(16) __bf16 As[2][BT_M][LD_S];   // M x K
  __shared__ __align__(16) __bf16 Bt[2][BT_N][LD_S];   // N x K (K-contig)

  const int tid  = threadIdx.x;
  const int lane = tid & 31;
  const int wave = tid >> 5;
  const int wr   = wave >> 2;   // 0..1
  const int wc   = wave & 3;    // 0..3
  const int m_blk = blockIdx.y * BT_M;
  const int n_blk = blockIdx.x * BT_N;

  // 4 async-load instructions per wave per stage.
  auto stage = [&](int buf, int kc) {
#pragma unroll
    for (int i = 0; i < 2; ++i) {
      int idx = tid + i * 256;
      int r = idx >> 2, c8 = idx & 3;
      copy16_g2s(&As[buf][r][c8 * 8],
                 xb + (size_t)(m_blk + r) * N_D + kc + c8 * 8);
      copy16_g2s(&Bt[buf][r][c8 * 8],
                 Wt + (size_t)(n_blk + r) * N_D + kc + c8 * 8);
    }
  };

  floatx8 acc[4][2];
#pragma unroll
  for (int i = 0; i < 4; ++i)
#pragma unroll
    for (int j = 0; j < 2; ++j) acc[i][j] = (floatx8){0,0,0,0,0,0,0,0};

  stage(0, 0);
  for (int kci = 0; kci < NK_STEPS; ++kci) {
    const int cur = kci & 1;
    __syncthreads();   // all waves done reading buf[cur^1] from prev iteration
    if (kci + 1 < NK_STEPS) {
      stage(cur ^ 1, (kci + 1) * BT_K);
      ASYNC_WAIT(4);   // in-order: current tile's 4 loads are complete
    } else {
      ASYNC_WAIT(0);
    }
    __syncthreads();

    bf16x16 af[4], bfv[2];
#pragma unroll
    for (int mt = 0; mt < 4; ++mt)
      af[mt] = frag_a(&As[cur][wr * 64 + mt * 16][0], LD_S, lane);
#pragma unroll
    for (int nt = 0; nt < 2; ++nt)
      bfv[nt] = frag_b(&Bt[cur][wc * 32 + nt * 16][0], LD_S, lane);
#pragma unroll
    for (int mt = 0; mt < 4; ++mt)
#pragma unroll
      for (int nt = 0; nt < 2; ++nt)
        acc[mt][nt] = wmma_bf16(af[mt], bfv[nt], acc[mt][nt]);
  }

  // Epilogue scatter.
  const int hi8 = (lane >> 4) << 3;
  const int n15 = lane & 15;
#pragma unroll
  for (int mt = 0; mt < 4; ++mt)
#pragma unroll
    for (int nt = 0; nt < 2; ++nt) {
      const int n_base = n_blk + wc * 32 + nt * 16;   // uniform per tile
      const int which  = n_base / N_D;                // 0=q, 1=k, 2=v
      const int n      = n_base + n15;
      const int d      = n - which * N_D;
      const int h      = d >> 6, di = d & 63;
      const float bv   = bias[n];
      const int m_base = m_blk + wr * 64 + mt * 16 + hi8;
      const int b      = m_base >> 11;
      const int t0     = m_base & (N_T - 1);
      if (which == 2) {
        // v transposed: [B,H,hd,T]; 8 consecutive t -> one 16B store.
        bf16x8 pk;
#pragma unroll
        for (int r = 0; r < 8; ++r) pk[r] = (__bf16)(acc[mt][nt][r] + bv);
        *(bf16x8*)(wv + ((size_t)(b * N_H + h) * N_HD + di) * N_T + t0) = pk;
      } else {
        __bf16* dst = (which == 0) ? wq : wk;         // [B,H,T,hd]
#pragma unroll
        for (int r = 0; r < 8; ++r)
          dst[((size_t)(b * N_H + h) * N_T + t0 + r) * N_HD + di] =
              (__bf16)(acc[mt][nt][r] + bv);
      }
    }
}

// ---------------------------------------------------------------------------
// Kernel 2: flash attention per (b,h, 64-row q tile). 128 threads = 4 waves.
// q,k: [B,H,T,hd]; v: [B,H,hd,T]. Double-buffered async K/V streaming.
// ---------------------------------------------------------------------------
#define LD_K 72   // 64 + 8 pad, multiple of 8

__global__ __launch_bounds__(128) void attn_kernel(
    const __bf16* __restrict__ q, const __bf16* __restrict__ k,
    const __bf16* __restrict__ v, __bf16* __restrict__ ao) {
  __shared__ __align__(16) __bf16 Ks[2][64][LD_K];    // key x hd
  __shared__ __align__(16) __bf16 Vt[2][64][LD_K];    // hd x key
  __shared__ __align__(16) __bf16 Ps[4][16][LD_K];    // per-wave P, row x key

  const int qt   = blockIdx.x;
  const int bh   = blockIdx.y;
  const int bidx = bh / N_H;
  const int h    = bh % N_H;
  const int tid  = threadIdx.x;
  const int lane = tid & 31;
  const int wave = tid >> 5;
  const int hi8  = (lane >> 4) << 3;
  const int n15  = lane & 15;
  const int qrow = qt * 64 + wave * 16;

  const __bf16* qb = q + (size_t)bh * N_T * N_HD;
  const __bf16* kbp = k + (size_t)bh * N_T * N_HD;
  const __bf16* vtp = v + (size_t)bh * N_HD * N_T;

  // 8 async-load instructions per wave per stage.
  auto stageKV = [&](int buf, int j) {
#pragma unroll
    for (int i = 0; i < 4; ++i) {
      int idx = tid + i * 128;
      int r = idx >> 3, c8 = idx & 7;
      copy16_g2s(&Ks[buf][r][c8 * 8],
                 kbp + (size_t)(j * 64 + r) * N_HD + c8 * 8);
      copy16_g2s(&Vt[buf][r][c8 * 8],
                 vtp + (size_t)r * N_T + j * 64 + c8 * 8);
    }
  };

  bf16x16 qa[2];
#pragma unroll
  for (int c = 0; c < 2; ++c)
    qa[c] = frag_a(qb + (size_t)qrow * N_HD + c * 32, N_HD, lane);

  floatx8 o[4];
#pragma unroll
  for (int i = 0; i < 4; ++i) o[i] = (floatx8){0,0,0,0,0,0,0,0};
  float mrow[8], lrow[8];
#pragma unroll
  for (int r = 0; r < 8; ++r) { mrow[r] = -1e30f; lrow[r] = 0.0f; }

  stageKV(0, 0);
  for (int j = 0; j <= qt; ++j) {
    const int cur = j & 1;
    __syncthreads();   // all waves done reading buf[cur^1] from prev iteration
    if (j + 1 <= qt) {
      stageKV(cur ^ 1, j + 1);
      ASYNC_WAIT(8);   // in-order: current K/V tile complete
    } else {
      ASYNC_WAIT(0);
    }
    __syncthreads();

    // S = Q @ K^T (16x64 per wave)
    floatx8 s[4];
#pragma unroll
    for (int tn = 0; tn < 4; ++tn) {
      floatx8 accs = (floatx8){0,0,0,0,0,0,0,0};
#pragma unroll
      for (int c = 0; c < 2; ++c)
        accs = wmma_bf16(qa[c], frag_b(&Ks[cur][tn * 16][c * 32], LD_K, lane),
                         accs);
      s[tn] = accs;
    }

    // Scale + causal mask on diagonal block.
    const bool diag = (j == qt);
#pragma unroll
    for (int tn = 0; tn < 4; ++tn)
#pragma unroll
      for (int r = 0; r < 8; ++r) {
        float val = s[tn][r] * 0.125f;
        if (diag && (j * 64 + tn * 16 + n15 > qrow + r + hi8)) val = -1e30f;
        s[tn][r] = val;
      }

    // Online softmax (row r+hi8 lives on 16 lanes of a half-wave).
#pragma unroll
    for (int r = 0; r < 8; ++r) {
      float mx = -1e30f;
#pragma unroll
      for (int tn = 0; tn < 4; ++tn) mx = fmaxf(mx, s[tn][r]);
#pragma unroll
      for (int off = 8; off > 0; off >>= 1)
        mx = fmaxf(mx, __shfl_xor(mx, off, 32));
      float mnew  = fmaxf(mrow[r], mx);
      float alpha = __expf(mrow[r] - mnew);
      mrow[r] = mnew;
      float ps = 0.0f;
#pragma unroll
      for (int tn = 0; tn < 4; ++tn) {
        float p = __expf(s[tn][r] - mnew);
        s[tn][r] = p;
        ps += p;
      }
#pragma unroll
      for (int off = 8; off > 0; off >>= 1) ps += __shfl_xor(ps, off, 32);
      lrow[r] = lrow[r] * alpha + ps;
#pragma unroll
      for (int tn = 0; tn < 4; ++tn) o[tn][r] *= alpha;
    }

    // Stage P into per-wave LDS (A-matrix row-major).
#pragma unroll
    for (int tn = 0; tn < 4; ++tn)
#pragma unroll
      for (int r = 0; r < 8; ++r)
        Ps[wave][r + hi8][tn * 16 + n15] = (__bf16)s[tn][r];
    __syncthreads();

    // O += P @ V
    bf16x16 pa[2];
#pragma unroll
    for (int c = 0; c < 2; ++c)
      pa[c] = frag_a(&Ps[wave][0][c * 32], LD_K, lane);
#pragma unroll
    for (int tn = 0; tn < 4; ++tn)
#pragma unroll
      for (int c = 0; c < 2; ++c)
        o[tn] = wmma_bf16(pa[c], frag_b(&Vt[cur][tn * 16][c * 32], LD_K, lane),
                          o[tn]);
  }

  // Normalize and write merged-head output [B,T,D] bf16.
#pragma unroll
  for (int tn = 0; tn < 4; ++tn)
#pragma unroll
    for (int r = 0; r < 8; ++r) {
      int trow = qrow + r + hi8;
      int col  = h * N_HD + tn * 16 + n15;
      ao[((size_t)bidx * N_T + trow) * N_D + col] = (__bf16)(o[tn][r] / lrow[r]);
    }
}

// ---------------------------------------------------------------------------
// Kernel 3: out = ao(bf16) @ Wot^T + b_o, fp32 output [8192,768].
// ---------------------------------------------------------------------------
__global__ __launch_bounds__(256) void oproj_gemm(
    const __bf16* __restrict__ A, const __bf16* __restrict__ Wt,
    const float* __restrict__ bias, float* __restrict__ out) {
  __shared__ __align__(16) __bf16 As[2][BT_M][LD_S];
  __shared__ __align__(16) __bf16 Bt[2][BT_N][LD_S];

  const int tid  = threadIdx.x;
  const int lane = tid & 31;
  const int wave = tid >> 5;
  const int wr   = wave >> 2;
  const int wc   = wave & 3;
  const int m_blk = blockIdx.y * BT_M;
  const int n_blk = blockIdx.x * BT_N;

  auto stage = [&](int buf, int kc) {
#pragma unroll
    for (int i = 0; i < 2; ++i) {
      int idx = tid + i * 256;
      int r = idx >> 2, c8 = idx & 3;
      copy16_g2s(&As[buf][r][c8 * 8],
                 A + (size_t)(m_blk + r) * N_D + kc + c8 * 8);
      copy16_g2s(&Bt[buf][r][c8 * 8],
                 Wt + (size_t)(n_blk + r) * N_D + kc + c8 * 8);
    }
  };

  floatx8 acc[4][2];
#pragma unroll
  for (int i = 0; i < 4; ++i)
#pragma unroll
    for (int j = 0; j < 2; ++j) acc[i][j] = (floatx8){0,0,0,0,0,0,0,0};

  stage(0, 0);
  for (int kci = 0; kci < NK_STEPS; ++kci) {
    const int cur = kci & 1;
    __syncthreads();
    if (kci + 1 < NK_STEPS) {
      stage(cur ^ 1, (kci + 1) * BT_K);
      ASYNC_WAIT(4);
    } else {
      ASYNC_WAIT(0);
    }
    __syncthreads();

    bf16x16 af[4], bfv[2];
#pragma unroll
    for (int mt = 0; mt < 4; ++mt)
      af[mt] = frag_a(&As[cur][wr * 64 + mt * 16][0], LD_S, lane);
#pragma unroll
    for (int nt = 0; nt < 2; ++nt)
      bfv[nt] = frag_b(&Bt[cur][wc * 32 + nt * 16][0], LD_S, lane);
#pragma unroll
    for (int mt = 0; mt < 4; ++mt)
#pragma unroll
      for (int nt = 0; nt < 2; ++nt)
        acc[mt][nt] = wmma_bf16(af[mt], bfv[nt], acc[mt][nt]);
  }

  const int hi8 = (lane >> 4) << 3;
  const int n15 = lane & 15;
#pragma unroll
  for (int mt = 0; mt < 4; ++mt)
#pragma unroll
    for (int nt = 0; nt < 2; ++nt) {
      int n = n_blk + wc * 32 + nt * 16 + n15;
      float bv = bias[n];
#pragma unroll
      for (int r = 0; r < 8; ++r) {
        int m = m_blk + wr * 64 + mt * 16 + r + hi8;
        out[(size_t)m * N_D + n] = acc[mt][nt][r] + bv;
      }
    }
}

// ---------------------------------------------------------------------------
extern "C" void kernel_launch(void* const* d_in, const int* in_sizes, int n_in,
                              void* d_out, int out_size, void* d_ws,
                              size_t ws_size, hipStream_t stream) {
  const float* x     = (const float*)d_in[0];
  const float* W_qkv = (const float*)d_in[1];
  const float* b_qkv = (const float*)d_in[2];
  const float* W_o   = (const float*)d_in[3];
  const float* b_o   = (const float*)d_in[4];
  float* out = (float*)d_out;

  // Workspace layout (bf16 elements):
  // xb | wq | wk | wv(transposed) | ao | Wqkvt | Wot
  __bf16* ws = (__bf16*)d_ws;
  const size_t per = (size_t)N_B * N_H * N_T * N_HD;   // 6,291,456
  __bf16* xb    = ws;
  __bf16* wq    = ws + per;
  __bf16* wk    = ws + 2 * per;
  __bf16* wv    = ws + 3 * per;
  __bf16* ao    = ws + 4 * per;
  __bf16* wqkvt = ws + 5 * per;                        // [2304][768]
  __bf16* wot   = wqkvt + (size_t)N_3D * N_D;          // [768][768]

  cvt_bf16_kernel<<<(N_M * N_D / 4 + 255) / 256, 256, 0, stream>>>(
      x, xb, N_M * N_D / 4);
  transpose_cvt_kernel<<<dim3(N_3D / 32, N_D / 32), 256, 0, stream>>>(
      W_qkv, wqkvt, N_D, N_3D);
  transpose_cvt_kernel<<<dim3(N_D / 32, N_D / 32), 256, 0, stream>>>(
      W_o, wot, N_D, N_D);

  qkv_gemm<<<dim3(N_3D / BT_N, N_M / BT_M), 256, 0, stream>>>(
      xb, wqkvt, b_qkv, wq, wk, wv);
  attn_kernel<<<dim3(N_T / 64, N_B * N_H), 128, 0, stream>>>(wq, wk, wv, ao);
  oproj_gemm<<<dim3(N_D / BT_N, N_M / BT_M), 256, 0, stream>>>(
      ao, wot, b_o, out);
}
